// ConformerBlock_16801912062469
// MI455X (gfx1250) — matrix operand order repaired and checked by reading, not compiled, not run on hardware
//
#include <hip/hip_runtime.h>

// ---------------------------------------------------------------------------
// Transformer block (pre-LN attn + FFN) for MI455X / gfx1250, bf16 WMMA path.
// ---------------------------------------------------------------------------

#define DEV __device__ __forceinline__

static constexpr int DIMC  = 1024;
static constexpr int HEADS = 16;
static constexpr int HD    = 64;
static constexpr int BB    = 2;
static constexpr int TT    = 2048;
static constexpr int BT    = BB * TT;   // 4096 rows

typedef __bf16 bf16;
typedef __attribute__((ext_vector_type(16))) __bf16 v16bf;
typedef __attribute__((ext_vector_type(8)))  float  v8f;

union FragC { v8f v; float f[8]; };

DEV bf16 f2bf(float x) {
  union { float f; unsigned u; } a; a.f = x;
  unsigned r = a.u + 0x7fffu + ((a.u >> 16) & 1u);   // round-to-nearest-even
  union { unsigned short s; bf16 b; } o;
  o.s = (unsigned short)(r >> 16);
  return o.b;
}

// A fragment, 16x32 bf16 (ISA 7.12.2): lanes 0-15 -> M, lane half selects K
// sub-bank; VGPR j holds K pair {.., ..}. All K offsets even -> dword loads.
DEV v16bf load_a_frag(const bf16* base, int lda, int lane) {
  const int m  = lane & 15;
  const int kh = lane >> 4;            // 0 or 1
  const bf16* p = base + m * lda + (kh << 3);
  union { v16bf v; unsigned u[8]; } f;
#pragma unroll
  for (int j = 0; j < 8; ++j) {
    const int k = ((j & 4) << 2) + ((j & 3) << 1);   // +16 for j>=4
    f.u[j] = *reinterpret_cast<const unsigned*>(p + k);
  }
  return f.v;
}

// B fragment, 32x16 bf16, tile stored n-major [n][ldb]: lanes 0-15 -> N,
// lane half h selects K 8h..8h+7 and 16+8h..16+8h+7 (same as A) -> 8 dword loads.
DEV v16bf load_b_frag(const bf16* base, int ldb, int lane) {
  const int n    = lane & 15;
  const int koff = (lane >> 4) << 3;
  const bf16* p = base + n * ldb + koff;
  union { v16bf v; unsigned u[8]; } f;
#pragma unroll
  for (int j = 0; j < 8; ++j)
    f.u[j] = *reinterpret_cast<const unsigned*>(p + ((j & 4) << 2) + ((j & 3) << 1));
  return f.v;
}

// ---------------------------------------------------------------------------
// fp32 -> bf16 elementwise convert (weights)
// ---------------------------------------------------------------------------
__global__ void cvt_kernel(const float* __restrict__ in, bf16* __restrict__ out, int n) {
  int idx = blockIdx.x * blockDim.x + threadIdx.x;
  const int stride = gridDim.x * blockDim.x;
  for (; idx < n; idx += stride) out[idx] = f2bf(in[idx]);
}

// ---------------------------------------------------------------------------
// LayerNorm over DIM=1024, one row per block (256 threads, 4 elems/thread)
// ---------------------------------------------------------------------------
__global__ __launch_bounds__(256)
void ln_kernel(const float* __restrict__ x, const float* __restrict__ g,
               const float* __restrict__ b, bf16* __restrict__ out) {
  __shared__ float s1[256];
  __shared__ float s2[256];
  const int row = blockIdx.x;
  const int t   = threadIdx.x;
  const float* xr = x + (size_t)row * DIMC;
  float a = 0.f, q = 0.f;
#pragma unroll
  for (int i = 0; i < 4; ++i) {
    const float v = xr[t + i * 256];
    a += v; q += v * v;
  }
  s1[t] = a; s2[t] = q;
  __syncthreads();
  for (int st = 128; st > 0; st >>= 1) {
    if (t < st) { s1[t] += s1[t + st]; s2[t] += s2[t + st]; }
    __syncthreads();
  }
  const float mean = s1[0] * (1.f / DIMC);
  const float var  = s2[0] * (1.f / DIMC) - mean * mean;
  const float rstd = rsqrtf(var + 1e-5f);
#pragma unroll
  for (int i = 0; i < 4; ++i) {
    const int idx = t + i * 256;
    out[(size_t)row * DIMC + idx] = f2bf((xr[idx] - mean) * rstd * g[idx] + b[idx]);
  }
}

// ---------------------------------------------------------------------------
// bf16 GEMM: C[M,N] = A[M,K] @ W[K,N] (+bias, +residual, +silu by MODE)
// BM=BN=128, BK=32. 256 threads = 8 waves, wave tile 32x64 (2x4 WMMA tiles).
// MODE 0: f32 out = acc+bias
// MODE 1: f32 out = acc+bias+res
// MODE 2: bf16 out = silu(acc+bias)
// ---------------------------------------------------------------------------
template <int MODE>
__global__ __launch_bounds__(256)
void gemm_bf16_kernel(const bf16* __restrict__ A, const bf16* __restrict__ W,
                      const float* __restrict__ bias, const float* __restrict__ res,
                      float* __restrict__ outF, bf16* __restrict__ outB,
                      int M, int N, int K) {
  constexpr int BM = 128, BN = 128, BK = 32;
  constexpr int LDA = BK + 2;   // 34 (even -> dword-aligned frag loads)
  constexpr int LDB = BK + 2;   // Bs stored n-major [n][k]
  __shared__ bf16 As[BM * LDA];
  __shared__ bf16 Bs[BN * LDB];

  const int tid  = threadIdx.x;
  const int lane = tid & 31;
  const int wave = tid >> 5;
  const int wm   = wave & 3;    // 4 waves along M
  const int wn   = wave >> 2;   // 2 waves along N
  const int m0   = blockIdx.y * BM;
  const int n0   = blockIdx.x * BN;

  FragC acc[2][4];
#pragma unroll
  for (int i = 0; i < 2; ++i)
#pragma unroll
    for (int j = 0; j < 4; ++j)
#pragma unroll
      for (int e = 0; e < 8; ++e) acc[i][j].f[e] = 0.f;

  const int arow = tid >> 2;           // 0..63  (and +64)
  const int acol = (tid & 3) << 3;     // 0,8,16,24
  const int brow = tid >> 3;           // k: 0..31
  const int bcol = (tid & 7) << 4;     // n: 0..112 step 16

  for (int k0 = 0; k0 < K; k0 += BK) {
    // --- stage A tile [m][k] ---
#pragma unroll
    for (int h = 0; h < 2; ++h) {
      const int r = arow + h * 64;
      const uint4 t4 = *reinterpret_cast<const uint4*>(A + (size_t)(m0 + r) * K + k0 + acol);
      unsigned* dst = reinterpret_cast<unsigned*>(&As[r * LDA + acol]);
      dst[0] = t4.x; dst[1] = t4.y; dst[2] = t4.z; dst[3] = t4.w;
    }
    // --- stage B tile, transposed to n-major [n][k] ---
    {
      const bf16* src = W + (size_t)(k0 + brow) * N + n0 + bcol;
      union { uint4 q; bf16 h[8]; } t0, t1;
      t0.q = *reinterpret_cast<const uint4*>(src);
      t1.q = *reinterpret_cast<const uint4*>(src + 8);
#pragma unroll
      for (int i2 = 0; i2 < 8; ++i2) {
        Bs[(bcol + i2) * LDB + brow]     = t0.h[i2];
        Bs[(bcol + 8 + i2) * LDB + brow] = t1.h[i2];
      }
    }
    if (k0 + BK < K) {   // global_prefetch_b8 for the next tiles
      __builtin_prefetch(A + (size_t)(m0 + arow) * K + k0 + BK + acol, 0, 0);
      __builtin_prefetch(W + (size_t)(k0 + BK + brow) * N + n0 + bcol, 0, 0);
    }
    __syncthreads();

    v16bf af[2], bfr[4];
#pragma unroll
    for (int i = 0; i < 2; ++i)
      af[i] = load_a_frag(&As[(wm * 32 + i * 16) * LDA], LDA, lane);
#pragma unroll
    for (int j = 0; j < 4; ++j)
      bfr[j] = load_b_frag(&Bs[(wn * 64 + j * 16) * LDB], LDB, lane);
#pragma unroll
    for (int i = 0; i < 2; ++i)
#pragma unroll
      for (int j = 0; j < 4; ++j)
        acc[i][j].v = __builtin_amdgcn_wmma_f32_16x16x32_bf16(
            false, af[i], false, bfr[j], (short)0, acc[i][j].v, false, false);
    __syncthreads();
  }

  // --- epilogue ---
  const int nl = lane & 15;
  const int mh = lane >> 4;
#pragma unroll
  for (int i = 0; i < 2; ++i)
#pragma unroll
    for (int j = 0; j < 4; ++j) {
      const int gcol = n0 + wn * 64 + j * 16 + nl;
      const float bv = bias[gcol];
#pragma unroll
      for (int r = 0; r < 8; ++r) {
        const int grow = m0 + wm * 32 + i * 16 + r + 8 * mh;
        float v = acc[i][j].f[r] + bv;
        if (MODE == 1) v += res[(size_t)grow * N + gcol];
        if (MODE == 2) {
          v = v / (1.f + __expf(-v));          // SiLU
          outB[(size_t)grow * N + gcol] = f2bf(v);
        } else {
          outF[(size_t)grow * N + gcol] = v;
        }
      }
    }
}

// ---------------------------------------------------------------------------
// RoPE split: fp32 qkv [BT, 3*DIM] -> bf16 q,k,v in [B,H,T,hd]
// ---------------------------------------------------------------------------
__global__ void rope_kernel(const float* __restrict__ qkv, bf16* __restrict__ q,
                            bf16* __restrict__ k, bf16* __restrict__ v) {
  const int d  = threadIdx.x;         // 0..63
  const int t  = blockIdx.x;          // 0..T-1
  const int bh = blockIdx.y;          // 0..B*H-1
  const int b  = bh >> 4;
  const int h  = bh & 15;
  const float* rp = qkv + ((size_t)b * TT + t) * (3 * DIMC);
  const int cq = h * HD + d;
  const float qv = rp[cq];
  const float kv = rp[DIMC + cq];
  const float vv = rp[2 * DIMC + cq];
  const int   dp   = (d < 32) ? d + 32 : d - 32;
  const float sign = (d < 32) ? -1.f : 1.f;
  const float qp = rp[h * HD + dp];
  const float kp = rp[DIMC + h * HD + dp];
  const int   i  = d & 31;
  const float inv = __expf(-0.28782314f * (float)i);   // 10000^(-i/32)
  const float ang = (float)t * inv;
  const float c = __cosf(ang), s = __sinf(ang);
  const size_t oidx = ((size_t)bh * TT + t) * HD + d;
  q[oidx] = f2bf(qv * c + sign * qp * s);
  k[oidx] = f2bf(kv * c + sign * kp * s);
  v[oidx] = f2bf(vv);
}

// ---------------------------------------------------------------------------
// Attention (non-causal flash style). Grid: (B*H, T/128), 256 threads.
// Each of 8 waves owns 16 query rows; block stages 64-key K/V tiles in LDS.
// ---------------------------------------------------------------------------
__global__ __launch_bounds__(256)
void attn_kernel(const bf16* __restrict__ Qp, const bf16* __restrict__ Kp,
                 const bf16* __restrict__ Vp, bf16* __restrict__ outA) {
  constexpr int LDK = HD + 2;               // 66
  __shared__ bf16 Ks[64 * LDK];             // [key][d]
  __shared__ bf16 Vs[HD * LDK];             // [d][key]   (n-major for P@V)
  __shared__ bf16 Ps[8 * 16 * LDK];         // per-wave P strip [16][64]

  const int tid  = threadIdx.x;
  const int lane = tid & 31;
  const int wave = tid >> 5;
  const int nl   = lane & 15;
  const int mh   = lane >> 4;
  const int bh   = blockIdx.x;
  const int qrow0 = blockIdx.y * 128 + wave * 16;
  const size_t base = (size_t)bh * TT * HD;
  const bf16* Qb = Qp + base;
  const bf16* Kb = Kp + base;
  const bf16* Vb = Vp + base;

  // q fragments (A layout), hd=64 -> two K=32 chunks, pinned in registers
  v16bf qf[2];
#pragma unroll
  for (int f = 0; f < 2; ++f)
    qf[f] = load_a_frag(Qb + (size_t)qrow0 * HD + f * 32, HD, lane);

  FragC o[4];
#pragma unroll
  for (int j = 0; j < 4; ++j)
#pragma unroll
    for (int e = 0; e < 8; ++e) o[j].f[e] = 0.f;
  float m_run[8], l_run[8];
#pragma unroll
  for (int r = 0; r < 8; ++r) { m_run[r] = -3.0e38f; l_run[r] = 0.f; }

  const int key = tid >> 2;            // 0..63
  const int dc  = (tid & 3) << 4;      // 0,16,32,48

  for (int kt = 0; kt < TT; kt += 64) {
    {  // K tile [key][d]
      const bf16* src = Kb + (size_t)(kt + key) * HD + dc;
      const uint4 a4 = *reinterpret_cast<const uint4*>(src);
      const uint4 b4 = *reinterpret_cast<const uint4*>(src + 8);
      unsigned* dst = reinterpret_cast<unsigned*>(&Ks[key * LDK + dc]);
      dst[0] = a4.x; dst[1] = a4.y; dst[2] = a4.z; dst[3] = a4.w;
      dst[4] = b4.x; dst[5] = b4.y; dst[6] = b4.z; dst[7] = b4.w;
    }
    {  // V tile transposed [d][key]
      const bf16* src = Vb + (size_t)(kt + key) * HD + dc;
      union { uint4 q; bf16 h[8]; } t0, t1;
      t0.q = *reinterpret_cast<const uint4*>(src);
      t1.q = *reinterpret_cast<const uint4*>(src + 8);
#pragma unroll
      for (int i2 = 0; i2 < 8; ++i2) {
        Vs[(dc + i2) * LDK + key]     = t0.h[i2];
        Vs[(dc + 8 + i2) * LDK + key] = t1.h[i2];
      }
    }
    __syncthreads();

    // S = q @ k^T   (16 x 64 per wave)
    FragC s[4];
#pragma unroll
    for (int nt = 0; nt < 4; ++nt) {
#pragma unroll
      for (int e = 0; e < 8; ++e) s[nt].f[e] = 0.f;
#pragma unroll
      for (int f = 0; f < 2; ++f)
        s[nt].v = __builtin_amdgcn_wmma_f32_16x16x32_bf16(
            false, qf[f], false,
            load_b_frag(&Ks[(nt * 16) * LDK + f * 32], LDK, lane),
            (short)0, s[nt].v, false, false);
    }

    // online softmax (rows across C-layout: row = r + 8*mh, N across 16 lanes)
#pragma unroll
    for (int r = 0; r < 8; ++r) {
      float mx = m_run[r];
#pragma unroll
      for (int nt = 0; nt < 4; ++nt) mx = fmaxf(mx, s[nt].f[r] * 0.125f);
#pragma unroll
      for (int off = 1; off < 16; off <<= 1) mx = fmaxf(mx, __shfl_xor(mx, off, 16));
      const float alpha = __expf(m_run[r] - mx);
      float rs = 0.f;
#pragma unroll
      for (int nt = 0; nt < 4; ++nt) {
        const float p = __expf(s[nt].f[r] * 0.125f - mx);
        Ps[wave * 16 * LDK + (r + 8 * mh) * LDK + nt * 16 + nl] = f2bf(p);
        rs += p;
      }
#pragma unroll
      for (int off = 1; off < 16; off <<= 1) rs += __shfl_xor(rs, off, 16);
      l_run[r] = l_run[r] * alpha + rs;
#pragma unroll
      for (int j = 0; j < 4; ++j) o[j].f[r] *= alpha;
      m_run[r] = mx;
    }
    asm volatile("s_wait_dscnt 0" ::: "memory");   // wave-private LDS RAW fence

    // O += P @ V
    v16bf pf[2];
#pragma unroll
    for (int f = 0; f < 2; ++f)
      pf[f] = load_a_frag(&Ps[wave * 16 * LDK + f * 32], LDK, lane);
#pragma unroll
    for (int ntd = 0; ntd < 4; ++ntd)
#pragma unroll
      for (int f = 0; f < 2; ++f)
        o[ntd].v = __builtin_amdgcn_wmma_f32_16x16x32_bf16(
            false, pf[f], false,
            load_b_frag(&Vs[(ntd * 16) * LDK + f * 32], LDK, lane),
            (short)0, o[ntd].v, false, false);
    __syncthreads();
  }

  // write [B,T,C] bf16 at column h*64
  const int bg = bh >> 4;
  const int h  = bh & 15;
#pragma unroll
  for (int ntd = 0; ntd < 4; ++ntd)
#pragma unroll
    for (int r = 0; r < 8; ++r) {
      const int qg  = qrow0 + r + 8 * mh;
      const int col = h * HD + ntd * 16 + nl;
      outA[((size_t)bg * TT + qg) * DIMC + col] = f2bf(o[ntd].f[r] / l_run[r]);
    }
}

// ---------------------------------------------------------------------------
// Host-side pipeline
// ---------------------------------------------------------------------------
extern "C" void kernel_launch(void* const* d_in, const int* in_sizes, int n_in,
                              void* d_out, int out_size, void* d_ws, size_t ws_size,
                              hipStream_t stream) {
  (void)in_sizes; (void)n_in; (void)out_size; (void)ws_size;
  const float* x      = (const float*)d_in[0];
  const float* ln1_g  = (const float*)d_in[1];
  const float* ln1_b  = (const float*)d_in[2];
  const float* W_qkv  = (const float*)d_in[3];
  const float* b_qkv  = (const float*)d_in[4];
  const float* W_proj = (const float*)d_in[5];
  const float* b_proj = (const float*)d_in[6];
  const float* ln2_g  = (const float*)d_in[7];
  const float* ln2_b  = (const float*)d_in[8];
  const float* W_ffn1 = (const float*)d_in[9];
  const float* b_ffn1 = (const float*)d_in[10];
  const float* W_ffn2 = (const float*)d_in[11];
  const float* b_ffn2 = (const float*)d_in[12];
  float* out = (float*)d_out;

  char* ws = (char*)d_ws;
  const size_t MB = 1ull << 20;
  bf16*  wqkv  = (bf16*)(ws + 0);         //  6 MB
  bf16*  wproj = (bf16*)(ws + 6 * MB);    //  2 MB
  bf16*  wffn1 = (bf16*)(ws + 8 * MB);    //  8 MB
  bf16*  wffn2 = (bf16*)(ws + 16 * MB);   //  8 MB
  bf16*  xn    = (bf16*)(ws + 24 * MB);   //  8 MB
  float* qkvf  = (float*)(ws + 32 * MB);  // 48 MB (freed after rope)
  bf16*  hbuf  = (bf16*)(ws + 32 * MB);   // 32 MB (reuses qkv region)
  bf16*  qb    = (bf16*)(ws + 80 * MB);   //  8 MB
  bf16*  kb    = (bf16*)(ws + 88 * MB);   //  8 MB
  bf16*  vb    = (bf16*)(ws + 96 * MB);   //  8 MB
  bf16*  attnb = (bf16*)(ws + 104 * MB);  //  8 MB
  float* x2    = (float*)(ws + 112 * MB); // 16 MB
  bf16*  xn2   = (bf16*)(ws + 128 * MB);  //  8 MB

  // weight conversion fp32 -> bf16
  cvt_kernel<<<512, 256, 0, stream>>>(W_qkv,  wqkv,  DIMC * 3 * DIMC);
  cvt_kernel<<<512, 256, 0, stream>>>(W_proj, wproj, DIMC * DIMC);
  cvt_kernel<<<512, 256, 0, stream>>>(W_ffn1, wffn1, DIMC * 4 * DIMC);
  cvt_kernel<<<512, 256, 0, stream>>>(W_ffn2, wffn2, 4 * DIMC * DIMC);

  // LN1
  ln_kernel<<<BT, 256, 0, stream>>>(x, ln1_g, ln1_b, xn);

  // QKV GEMM (fp32 out for RoPE)
  gemm_bf16_kernel<0><<<dim3(3 * DIMC / 128, BT / 128), 256, 0, stream>>>(
      xn, wqkv, b_qkv, nullptr, qkvf, nullptr, BT, 3 * DIMC, DIMC);

  // RoPE + head split
  rope_kernel<<<dim3(TT, BB * HEADS), 64, 0, stream>>>(qkvf, qb, kb, vb);

  // attention
  attn_kernel<<<dim3(BB * HEADS, TT / 128), 256, 0, stream>>>(qb, kb, vb, attnb);

  // proj GEMM + residual x
  gemm_bf16_kernel<1><<<dim3(DIMC / 128, BT / 128), 256, 0, stream>>>(
      attnb, wproj, b_proj, x, x2, nullptr, BT, DIMC, DIMC);

  // LN2
  ln_kernel<<<BT, 256, 0, stream>>>(x2, ln2_g, ln2_b, xn2);

  // FFN1 GEMM + SiLU (bf16 out)
  gemm_bf16_kernel<2><<<dim3(4 * DIMC / 128, BT / 128), 256, 0, stream>>>(
      xn2, wffn1, b_ffn1, nullptr, nullptr, hbuf, BT, 4 * DIMC, DIMC);

  // FFN2 GEMM + residual x2 -> d_out
  gemm_bf16_kernel<1><<<dim3(DIMC / 128, BT / 128), 256, 0, stream>>>(
      hbuf, wffn2, b_ffn2, x2, out, nullptr, BT, DIMC, 4 * DIMC);
}
